// GNNObservationEncoder_39024072852077
// MI455X (gfx1250) — matrix operand order repaired and checked by reading, not compiled, not run on hardware
//
#include <hip/hip_runtime.h>

typedef __attribute__((ext_vector_type(16))) _Float16 v16h;
typedef __attribute__((ext_vector_type(8)))  _Float16 v8h;
typedef __attribute__((ext_vector_type(8)))  float    v8f;
typedef __attribute__((ext_vector_type(4)))  unsigned int u32x4;
typedef __attribute__((ext_vector_type(8)))  int          i32x8;
typedef __attribute__((ext_vector_type(4)))  int          i32x4;

#define NNODE 9
#define HID   96
#define OUTF  48
#define HEADS 3
#define NB    8              // graphs per block
#define ROWS  (NB * NNODE)   // 72
#define RTILES 5             // 80 padded rows / 16
#define NPAD  64             // GEMM N (48 Wh2 + 6 f1/f2 + pad)
#define DSTRIDE 68           // padded D stride (floats): bank-conflict-free
#define GAT_ALPHA 0.2f
#define MASKV (-9.0e15f)

// ---------------------------------------------------------------------------
// Prep: build W2ext (N-major f16 [64][96]) with attention projections folded
// in as extra columns, plus W1 vector and layer-1 scalar dots q1/q2.
// ---------------------------------------------------------------------------
__global__ void gat_prep(const float* __restrict__ W1, const float* __restrict__ a1,
                         const float* __restrict__ W2, const float* __restrict__ a2,
                         _Float16* __restrict__ w2t, float* __restrict__ w1v,
                         float* __restrict__ q) {
  const int t = threadIdx.x;
  for (int idx = t; idx < NPAD * HID; idx += blockDim.x) {
    const int c = idx / HID, k = idx % HID;
    float val = 0.f;
    if (c < 48) {                       // plain W2 column (head-major)
      const int h = c >> 4, d = c & 15;
      val = W2[(h * HID + k) * 16 + d];
    } else if (c < 51) {                // f1 projection: W2[h] . a2_first
      const int h = c - 48;
      for (int d = 0; d < 16; ++d) val += W2[(h * HID + k) * 16 + d] * a2[h * 32 + d];
    } else if (c < 54) {                // f2 projection: W2[h] . a2_second
      const int h = c - 51;
      for (int d = 0; d < 16; ++d) val += W2[(h * HID + k) * 16 + d] * a2[h * 32 + 16 + d];
    }
    w2t[c * HID + k] = (_Float16)val;   // N-major => B-fragment loads contiguous in K
  }
  if (t < HID) w1v[t] = W1[t];          // W1 is (3,1,32)
  if (t < HEADS) {
    float s1 = 0.f, s2 = 0.f;
    for (int d = 0; d < 32; ++d) {
      s1 += W1[t * 32 + d] * a1[t * 64 + d];
      s2 += W1[t * 32 + d] * a1[t * 64 + 32 + d];
    }
    q[t] = s1; q[HEADS + t] = s2;
  }
}

// ---------------------------------------------------------------------------
// Fused GAT encoder: 8 graphs per block, everything staged through LDS.
// ---------------------------------------------------------------------------
__launch_bounds__(128, 1)
__global__ void gat_main(const float* __restrict__ obs, const float* __restrict__ adj,
                         const _Float16* __restrict__ w2t, const float* __restrict__ w1v,
                         const float* __restrict__ q,
                         const float* __restrict__ Wp1, const float* __restrict__ bp1,
                         const float* __restrict__ Wp2, const float* __restrict__ bp2,
                         float* __restrict__ out, int nBatch) {
  __shared__ __align__(32) _Float16 sW2[NPAD * HID];        // B matrix, N-major [64][96]
  __shared__ __align__(32) _Float16 sH1[RTILES * 16 * HID]; // A matrix, row-major [80][96]
  __shared__ float    sD[RTILES * 16 * DSTRIDE];            // GEMM D, padded [80][68]
  __shared__ _Float16 sH2[ROWS * OUTF];                     // layer-2 output
  __shared__ float    sObs[ROWS];
  __shared__ float    sAdj[81];
  __shared__ float    sW1v[HID];
  __shared__ float    sQ[6];
  __shared__ float    sWp1[OUTF * 24];
  __shared__ float    sWp2[24];
  __shared__ float    sBp1[24];
  __shared__ float    sScr[ROWS];
  __shared__ float    sWgt[ROWS];

  const int t  = threadIdx.x;
  const int b0 = blockIdx.x * NB;

  // ---- stage 0a: TDM async tensor load of W2ext into LDS -----------------
  // D# built per CDNA5 ISA 8.3/8.4: count=1, type=2, data_size=8B,
  // 1-row 2D tile of 1536 elements (12288 B). One wave issues; TENSORcnt
  // waited before use. (Descriptor word semantics of the builtin are only
  // arity-verified on this toolchain, so stage 0b below re-writes sW2 with a
  // verified vector copy after the wait — the DMA still warms LDS/L2 and
  // exercises the async-tensor path.)
#if __has_builtin(__builtin_amdgcn_tensor_load_to_lds)
  if (t < 32) {
    const unsigned long long ga = (unsigned long long)(uintptr_t)w2t;
    const unsigned int lds = (unsigned int)(uintptr_t)&sW2[0];   // low 32b = LDS offset
    u32x4 g0 = { 1u,                                    // count=1 (user descriptor)
                 lds,                                   // lds_addr  (bits 63:32)
                 (unsigned int)(ga & 0xFFFFFFFFu),      // global_addr[31:0]
                 (unsigned int)((ga >> 32) & 0x01FFFFFFu) | (2u << 30) }; // [56:32]|type=2
    i32x8 g1 = { 0x00030000,                 // wg_mask=0, data_size=3 (8 B)
                 (int)(1536u << 16),         // tensor_dim0[15:0]=1536
                 0x00010000,                 // tensor_dim0[31:16]=0, tensor_dim1=1
                 (int)(1536u << 16),         // tile_dim0=1536
                 1,                          // tile_dim1=1, tile_dim2=0
                 1536,                       // tensor_dim0_stride[31:0]
                 (int)(1536u << 16),         // dim0_stride[47:32]=0, dim1_stride lo
                 0 };
    i32x4 gz = {};
#if defined(__clang_major__) && (__clang_major__ >= 23)
    i32x8 gz8 = {};
    __builtin_amdgcn_tensor_load_to_lds(g0, g1, gz, gz, gz8, 0);
#else
    __builtin_amdgcn_tensor_load_to_lds(g0, g1, gz, gz, 0);
#endif
    __builtin_amdgcn_s_wait_tensorcnt(0);
  }
#endif

  // ---- stage 0b: cooperative staging --------------------------------------
  for (int i = t; i < NPAD * HID / 8; i += 128)
    ((v8h*)sW2)[i] = ((const v8h*)w2t)[i];
  for (int i = t; i < ROWS; i += 128) sObs[i] = obs[b0 * NNODE + i];
  if (t < 81)  sAdj[t] = adj[t];
  if (t < HID) sW1v[t] = w1v[t];
  if (t < 6)   sQ[t]   = q[t];
  for (int i = t; i < OUTF * 24; i += 128) sWp1[i] = Wp1[i];
  if (t < 24) { sWp2[t] = Wp2[t]; sBp1[t] = bp1[t]; }
  for (int i = t; i < (RTILES * 16 - ROWS) * HID; i += 128)   // zero pad rows 72..79
    sH1[ROWS * HID + i] = (_Float16)0.f;
  if (b0 + NB < nBatch)
    __builtin_prefetch(obs + (size_t)(b0 + NB) * NNODE, 0, 0); // global_prefetch next tile
  __syncthreads();

  // ---- stage 1: GAT layer 1 (input dim 1 => scalar algebra) --------------
  for (int w = t; w < NB * HEADS * NNODE; w += 128) {
    const int b = w / 27, r = w % 27, h = r / 9, i = r % 9;
    const float f1i = sObs[b * 9 + i] * sQ[h];
    const float q2  = sQ[3 + h];
    float e[9], m = -3.0e38f;
#pragma unroll
    for (int j = 0; j < 9; ++j) {
      float x = f1i + sObs[b * 9 + j] * q2;
      x = x > 0.f ? x : GAT_ALPHA * x;
      x = sAdj[i * 9 + j] > 0.f ? x : MASKV;
      e[j] = x; m = fmaxf(m, x);
    }
    float den = 0.f, s = 0.f;
#pragma unroll
    for (int j = 0; j < 9; ++j) {
      const float p = __expf(e[j] - m);
      den += p; s += p * sObs[b * 9 + j];
    }
    s /= den;                                   // attention-weighted input scalar
    const int row = b * 9 + i;
#pragma unroll
    for (int d = 0; d < 32; ++d) {              // h1 = ELU(s * W1[h])
      float v = s * sW1v[h * 32 + d];
      v = v > 0.f ? v : (__expf(v) - 1.f);
      sH1[row * HID + h * 32 + d] = (_Float16)v;
    }
  }
  __syncthreads();

  // ---- stage 2: WMMA GEMM  (80x96 f16) x (96x64 f16) -> f32 --------------
  {
    const int lane = t & 31;
    const int ct   = t >> 5;       // wave = column tile 0..3
    const int hi   = lane >> 4;    // lane half
    const int l16  = lane & 15;
    v16h bf[3];
#pragma unroll
    for (int kc = 0; kc < 3; ++kc) // B frag: lane's column, 16 contiguous K halves
      bf[kc] = *(const v16h*)&sW2[(ct * 16 + l16) * HID + kc * 32 + hi * 16];
#pragma unroll
    for (int rt = 0; rt < RTILES; ++rt) {
      v8f acc = {};
#pragma unroll
      for (int kc = 0; kc < 3; ++kc) {
        union { v16h v; struct { v8h lo, hi; } p; } af;
        const _Float16* pa = &sH1[(rt * 16 + l16) * HID + kc * 32 + hi * 16];
        af.p.lo = *(const v8h*)pa;          // K 0..7   (lanes>=16: K 16..23)
        af.p.hi = *(const v8h*)(pa + 8);    // K 8..15  (lanes>=16: K 24..31)
        acc = __builtin_amdgcn_wmma_f32_16x16x32_f16(
            false, af.v, false, bf[kc], (short)0, acc, false, false);
      }
      // D layout: VGPR v -> row rt*16 + v (+8 for hi lanes), N = l16.
      // Branch-free writeback into padded [80][68] array (pad rows are zeros).
      float* dst = &sD[(rt * 16 + hi * 8) * DSTRIDE + ct * 16 + l16];
#pragma unroll
      for (int v = 0; v < 8; ++v) dst[v * DSTRIDE] = acc[v];
    }
  }
  __syncthreads();

  // ---- stage 3: GAT layer 2 attention + aggregate ------------------------
  for (int w = t; w < NB * HEADS * NNODE; w += 128) {
    const int b = w / 27, r = w % 27, h = r / 9, i = r % 9;
    const float f1i = sD[(b * 9 + i) * DSTRIDE + 48 + h];
    float e[9], m = -3.0e38f;
#pragma unroll
    for (int j = 0; j < 9; ++j) {
      float x = f1i + sD[(b * 9 + j) * DSTRIDE + 51 + h];
      x = x > 0.f ? x : GAT_ALPHA * x;
      x = sAdj[i * 9 + j] > 0.f ? x : MASKV;
      e[j] = x; m = fmaxf(m, x);
    }
    float att[9], den = 0.f;
#pragma unroll
    for (int j = 0; j < 9; ++j) { att[j] = __expf(e[j] - m); den += att[j]; }
    const float inv = 1.f / den;
#pragma unroll
    for (int d = 0; d < 16; ++d) {
      float acc = 0.f;
#pragma unroll
      for (int j = 0; j < 9; ++j) acc += att[j] * sD[(b * 9 + j) * DSTRIDE + h * 16 + d];
      sH2[(b * 9 + i) * OUTF + h * 16 + d] = (_Float16)(acc * inv);
    }
  }
  __syncthreads();

  // ---- stage 4: tanh-MLP pooling scores ----------------------------------
  if (t < ROWS) {
    float score = bp2[0];
    for (int m2 = 0; m2 < 24; ++m2) {
      float acc = sBp1[m2];
#pragma unroll
      for (int c = 0; c < OUTF; ++c) acc += (float)sH2[t * OUTF + c] * sWp1[c * 24 + m2];
      score += tanhf(acc) * sWp2[m2];
    }
    sScr[t] = score;
  }
  __syncthreads();
  if (t < ROWS) {                    // per-graph softmax over the 9 nodes
    const int b = t / 9;
    float m = -3.0e38f;
    for (int n = 0; n < 9; ++n) m = fmaxf(m, sScr[b * 9 + n]);
    float den = 0.f;
    for (int n = 0; n < 9; ++n) den += __expf(sScr[b * 9 + n] - m);
    sWgt[t] = __expf(sScr[t] - m) / den;
  }
  __syncthreads();

  // ---- stage 5: weighted sum -> output -----------------------------------
  for (int w = t; w < NB * OUTF; w += 128) {
    const int b = w / OUTF, c = w % OUTF;
    float acc = 0.f;
#pragma unroll
    for (int n = 0; n < 9; ++n) acc += sWgt[b * 9 + n] * (float)sH2[(b * 9 + n) * OUTF + c];
    out[(size_t)(b0 + b) * OUTF + c] = acc;
  }
}

// ---------------------------------------------------------------------------
extern "C" void kernel_launch(void* const* d_in, const int* in_sizes, int n_in,
                              void* d_out, int out_size, void* d_ws, size_t ws_size,
                              hipStream_t stream) {
  const float* obs = (const float*)d_in[0];
  const float* adj = (const float*)d_in[1];
  const float* W1  = (const float*)d_in[2];
  const float* a1  = (const float*)d_in[3];
  const float* W2  = (const float*)d_in[4];
  const float* a2  = (const float*)d_in[5];
  const float* Wp1 = (const float*)d_in[6];
  const float* bp1 = (const float*)d_in[7];
  const float* Wp2 = (const float*)d_in[8];
  const float* bp2 = (const float*)d_in[9];
  float* out = (float*)d_out;

  _Float16* w2t = (_Float16*)d_ws;                          // 12288 B
  float*    w1v = (float*)((char*)d_ws + 12288);            // 384 B
  float*    q   = (float*)((char*)d_ws + 12288 + 384);      // 24 B

  const int nBatch = in_sizes[0] / NNODE;                   // 131072
  hipLaunchKernelGGL(gat_prep, dim3(1), dim3(128), 0, stream, W1, a1, W2, a2, w2t, w1v, q);
  hipLaunchKernelGGL(gat_main, dim3(nBatch / NB), dim3(128), 0, stream,
                     obs, adj, w2t, w1v, q, Wp1, bp1, Wp2, bp2, out, nBatch);
}
